// PCLocEncoder_80900003987591
// MI455X (gfx1250) — matrix operand order repaired and checked by reading, not compiled, not run on hardware
//
#include <hip/hip_runtime.h>
#include <hip/hip_bf16.h>
#include <math.h>

typedef _Float16 v16h __attribute__((ext_vector_type(16)));
typedef _Float16 v8h  __attribute__((ext_vector_type(8)));
typedef float    v8f  __attribute__((ext_vector_type(8)));
typedef unsigned int u32x4 __attribute__((ext_vector_type(4)));
typedef int          i32x4 __attribute__((ext_vector_type(4)));
typedef int          i32x8 __attribute__((ext_vector_type(8)));

// ---------------------------------------------------------------------------
// Farthest point sampling. One workgroup per batch; running distance array in
// LDS (N <= 8192 -> 32KB of the 320KB WGP LDS). Reproduces the jax scan:
// out[0] = 0, then out[i] = argmax(distance) after i min-updates.
// ---------------------------------------------------------------------------
__global__ void fps_kernel(const float* __restrict__ xyz, int N, int npoint,
                           int* __restrict__ out) {
  __shared__ float dist[8192];
  __shared__ float red_v[256];
  __shared__ int   red_i[256];
  const int b = blockIdx.x;
  const int t = threadIdx.x;
  const float* P = xyz + (size_t)b * N * 3;
  for (int i = t; i < N; i += 256) dist[i] = 1e10f;
  __syncthreads();
  int far = 0;
  for (int s = 0; s < npoint; ++s) {
    if (t == 0) out[b * npoint + s] = far;
    const float cx = P[far * 3 + 0];
    const float cy = P[far * 3 + 1];
    const float cz = P[far * 3 + 2];
    float best = -1.0f; int bi = 0;
    for (int i = t; i < N; i += 256) {
      const float dx = P[i * 3 + 0] - cx;
      const float dy = P[i * 3 + 1] - cy;
      const float dz = P[i * 3 + 2] - cz;
      const float d  = dx * dx + dy * dy + dz * dz;
      const float nd = fminf(dist[i], d);
      dist[i] = nd;
      if (nd > best) { best = nd; bi = i; }
    }
    red_v[t] = best; red_i[t] = bi;
    __syncthreads();
    for (int off = 128; off; off >>= 1) {
      if (t < off) {
        if (red_v[t + off] > red_v[t] ||
            (red_v[t + off] == red_v[t] && red_i[t + off] < red_i[t])) {
          red_v[t] = red_v[t + off];
          red_i[t] = red_i[t + off];
        }
      }
      __syncthreads();
    }
    far = red_i[0];
    __syncthreads();
  }
}

// Gather sampled coordinates: new_xyz[b,s,:] = xyz[b, fidx[b,s], :]
__global__ void gather_xyz_kernel(const float* __restrict__ xyz,
                                  const int* __restrict__ fidx,
                                  float* __restrict__ out, int N, int S, int B) {
  const int i = blockIdx.x * blockDim.x + threadIdx.x;
  if (i >= B * S) return;
  const int b = i / S;
  const int p = fidx[i];
  const float* src = xyz + ((size_t)b * N + p) * 3;
  float* dst = out + (size_t)i * 3;
  dst[0] = src[0]; dst[1] = src[1]; dst[2] = src[2];
}

// ---------------------------------------------------------------------------
// Ball query. One wave32 per query point. Collect first `nsample` in-radius
// indices in ascending order via ballot+popc prefix; pad with the first hit.
// ---------------------------------------------------------------------------
__global__ void ballq_kernel(const float* __restrict__ xyz,
                             const float* __restrict__ new_xyz,
                             int* __restrict__ idx,
                             int B, int N, int S, int nsample, float r2) {
  const int wave = (blockIdx.x * blockDim.x + threadIdx.x) >> 5;
  const int lane = threadIdx.x & 31;
  if (wave >= B * S) return;
  const int b = wave / S;
  const float* P = xyz + (size_t)b * N * 3;
  const float qx = new_xyz[(size_t)wave * 3 + 0];
  const float qy = new_xyz[(size_t)wave * 3 + 1];
  const float qz = new_xyz[(size_t)wave * 3 + 2];
  int* out = idx + (size_t)wave * nsample;
  int cnt = 0, first = -1;
  for (int base = 0; base < N && cnt < nsample; base += 32) {
    const int i = base + lane;
    bool ok = false;
    if (i < N) {
      const float dx = P[i * 3 + 0] - qx;
      const float dy = P[i * 3 + 1] - qy;
      const float dz = P[i * 3 + 2] - qz;
      ok = (dx * dx + dy * dy + dz * dz) <= r2;
    }
    const unsigned m = (unsigned)__ballot(ok);
    if (first < 0 && m) first = base + __builtin_ctz(m);
    const int pos = cnt + __popc(m & ((1u << lane) - 1u));
    if (ok && pos < nsample) out[pos] = i;
    cnt += __popc(m);
  }
  if (cnt < nsample) {
    for (int p = cnt + lane; p < nsample; p += 32) out[p] = first;
  }
}

// ---------------------------------------------------------------------------
// Grouping: build f16 activation rows [B*S*K x Kpad], channels 0..2 are
// (xyz[idx] - new_xyz), 3..3+Cp-1 are gathered point features, rest zero-pad.
// ---------------------------------------------------------------------------
__global__ void group_rel_kernel(const float* __restrict__ xyz,
                                 const float* __restrict__ new_xyz,
                                 const int* __restrict__ idx,
                                 const _Float16* __restrict__ pts, int Cp,
                                 _Float16* __restrict__ out,
                                 int N, int S, int K, int Kpad, int total) {
  const int i = blockIdx.x * blockDim.x + threadIdx.x;
  if (i >= total) return;
  const int c  = i % Kpad;
  const int g  = i / Kpad;               // (b,s,k) flat
  const int sq = g / K;                  // (b,s) flat
  const int b  = sq / S;
  const int pi = idx[g];
  _Float16 v = (_Float16)0.0f;
  if (c < 3) {
    v = (_Float16)(xyz[((size_t)b * N + pi) * 3 + c] - new_xyz[(size_t)sq * 3 + c]);
  } else if (pts != nullptr && c < 3 + Cp) {
    v = pts[((size_t)b * N + pi) * Cp + (c - 3)];
  }
  out[i] = v;
}

// group_all: rows = (b, s), channels [raw xyz | pts | pad]
__global__ void group_all_kernel(const float* __restrict__ xyz,
                                 const _Float16* __restrict__ pts, int Cp,
                                 _Float16* __restrict__ out, int Kpad, int total) {
  const int i = blockIdx.x * blockDim.x + threadIdx.x;
  if (i >= total) return;
  const int c = i % Kpad;
  const int r = i / Kpad;
  _Float16 v = (_Float16)0.0f;
  if (c < 3)               v = (_Float16)xyz[(size_t)r * 3 + c];
  else if (c < 3 + Cp)     v = pts[(size_t)r * Cp + (c - 3)];
  out[i] = v;
}

// ---------------------------------------------------------------------------
// Per-layer prep: convert W (O x C, f32) to f16 zero-padded [O x Kpad], and
// fold BN into per-channel affine: y = scale*(Wx) + shift, relu.
// ---------------------------------------------------------------------------
__global__ void prep_layer_kernel(const float* __restrict__ W,
                                  const float* __restrict__ b,
                                  const float* __restrict__ g,
                                  const float* __restrict__ bt,
                                  const float* __restrict__ mu,
                                  const float* __restrict__ var,
                                  int O, int C, int Kpad,
                                  _Float16* __restrict__ Wh,
                                  float* __restrict__ scale,
                                  float* __restrict__ shift) {
  const int i = blockIdx.x * blockDim.x + threadIdx.x;
  if (i < O * Kpad) {
    const int o = i / Kpad, k = i % Kpad;
    Wh[i] = (k < C) ? (_Float16)W[(size_t)o * C + k] : (_Float16)0.0f;
  }
  if (i < O) {
    const float sc = g[i] * rsqrtf(var[i] + 1e-5f);
    scale[i] = sc;
    shift[i] = sc * (b[i] - mu[i]) + bt[i];
  }
}

// ---------------------------------------------------------------------------
// WMMA GEMM + fused BN + ReLU. One wave32 per 16 x (16*NT) output stripe.
//   out[m,n] = relu( scale[n]*sum_k A[m,k]*Wh[n,k] + shift[n] ),  f16 out.
// The weight stripe [NT*16 x Kpad] is staged into LDS once per block — via the
// Tensor Data Mover (tensor_load_to_lds + s_wait_tensorcnt) when the builtin
// is available, else a vectorized copy — then read back with ds loads, so the
// K loop issues NT independent v_wmma_f32_16x16x32_f16 per step with A from
// global (prefetched) and B from LDS.
// ---------------------------------------------------------------------------
template <int NT>
__global__ void wmma_gemm_bn_relu(const _Float16* __restrict__ A,
                                  const _Float16* __restrict__ Wh,
                                  const float* __restrict__ scale,
                                  const float* __restrict__ shift,
                                  _Float16* __restrict__ out,
                                  int M, int Kpad, int O) {
  extern __shared__ v16h lds_v[];                 // 32B-aligned LDS stripe
  _Float16* ldsB = (_Float16*)lds_v;              // [NT*16 rows][Kpad] halves
  const int lane  = threadIdx.x;                  // blockDim.x == 32
  const int half_ = lane >> 4;                    // 0 or 1
  const int m0 = blockIdx.x * 16;
  const int n0 = blockIdx.y * (16 * NT);
  const int row = m0 + (lane & 15);

  const _Float16* gB = Wh + (size_t)n0 * Kpad;    // weight stripe start
  const int stripeHalves = NT * 16 * Kpad;

#if __has_builtin(__builtin_amdgcn_tensor_load_to_lds)
  {
    // 2D TDM descriptor (CDNA5 ISA ch.8): tile = [NT*16 rows x Kpad cols],
    // 2-byte elements, row stride = Kpad.
    const unsigned long long ga = (unsigned long long)(uintptr_t)gB;
    const unsigned lds_addr = (unsigned)(uintptr_t)ldsB;   // flat LDS low bits
    const unsigned kp = (unsigned)Kpad;
    const unsigned rows = (unsigned)(NT * 16);
    u32x4 g0 = {1u,                                   // count=1, user desc
                lds_addr,                             // lds_addr
                (unsigned)(ga & 0xffffffffu),         // global_addr[31:0]
                (unsigned)((ga >> 32) & 0x01ffffffu) | (2u << 30)}; // [56:32] | type=2
    i32x8 g1 = {(int)(1u << 16),                      // wg_mask=0, data_size=2B
                (int)((kp & 0xffffu) << 16),          // tensor_dim0[15:0]
                (int)((kp >> 16) | (rows << 16)),     // dim0[31:16] | tensor_dim1[15:0]
                (int)(kp << 16),                      // dim1[31:16]=0 | tile_dim0=Kpad
                (int)rows,                            // tile_dim1 | tile_dim2=0
                (int)kp,                              // tensor_dim0_stride[31:0]
                0, 0};                                // stride hi / dim1_stride = 0
    i32x4 z4 = {0, 0, 0, 0};
#if defined(__clang_major__) && (__clang_major__ >= 23)
    i32x8 z8 = {0, 0, 0, 0, 0, 0, 0, 0};
    __builtin_amdgcn_tensor_load_to_lds(g0, g1, z4, z4, z8, 0);
#else
    __builtin_amdgcn_tensor_load_to_lds(g0, g1, z4, z4, 0);
#endif
  }
#if __has_builtin(__builtin_amdgcn_s_wait_tensorcnt)
  __builtin_amdgcn_s_wait_tensorcnt(0);
#else
  asm volatile("s_wait_tensorcnt 0x0" ::: "memory");
#endif
#else
  // Fallback: vectorized cooperative copy (stripeHalves is a multiple of 256).
  for (int i = lane * 8; i < stripeHalves; i += 32 * 8) {
    *(v8h*)(ldsB + i) = *(const v8h*)(gB + i);
  }
#endif
  __syncthreads();   // single-wave WG: compiler memory fence for LDS

  // A operand pointers (16 contiguous K at half*16, matching B).
  const _Float16* aptr = A + (size_t)row * Kpad + half_ * 16;
  // Per-tile LDS B pointers (B operand: 16 contiguous K at half*16, same k order as A).
  const _Float16* bptr[NT];
#pragma unroll
  for (int t = 0; t < NT; ++t)
    bptr[t] = ldsB + (size_t)((lane & 15) + t * 16) * Kpad + half_ * 16;

  v8f acc[NT];
#pragma unroll
  for (int t = 0; t < NT; ++t) acc[t] = (v8f){};

  for (int k0 = 0; k0 < Kpad; k0 += 32) {
    if (k0 + 32 < Kpad) {
      __builtin_prefetch((const void*)(aptr + k0 + 32), 0, 1);  // global_prefetch_b8
    }
    const v8h lo = *(const v8h*)(aptr + k0);
    const v8h hi = *(const v8h*)(aptr + k0 + 8);
    const v16h a = __builtin_shufflevector(lo, hi, 0, 1, 2, 3, 4, 5, 6, 7,
                                                   8, 9, 10, 11, 12, 13, 14, 15);
#pragma unroll
    for (int t = 0; t < NT; ++t) {
      const v16h bm = *(const v16h*)(bptr[t] + k0);   // ds_load from LDS
      acc[t] = __builtin_amdgcn_wmma_f32_16x16x32_f16(
          false, a, false, bm, (short)0, acc[t], false, false);
    }
  }

#pragma unroll
  for (int t = 0; t < NT; ++t) {
    const int ncol = n0 + t * 16 + (lane & 15);
    const float sc = scale[ncol];
    const float sh = shift[ncol];
#pragma unroll
    for (int j = 0; j < 8; ++j) {
      float r = acc[t][j] * sc + sh;
      r = fmaxf(r, 0.0f);
      const int rrow = m0 + j + half_ * 8;     // C/D layout: VGPR j -> M=j(+8)
      out[(size_t)rrow * O + ncol] = (_Float16)r;
    }
  }
}

// Max-pool over the group (K) dimension. out is f16 (feeds next stage) or f32
// (final network output).
__global__ void maxpool_kernel(const _Float16* __restrict__ in, int Kg, int O,
                               int Stotal, _Float16* __restrict__ out_h,
                               float* __restrict__ out_f) {
  const int i = blockIdx.x * blockDim.x + threadIdx.x;
  if (i >= Stotal * O) return;
  const int s = i / O, c = i % O;
  const _Float16* p = in + ((size_t)s * Kg) * O + c;
  float m = -3.0e38f;
  for (int k = 0; k < Kg; ++k) m = fmaxf(m, (float)p[(size_t)k * O]);
  if (out_h) out_h[i] = (_Float16)m;
  if (out_f) out_f[i] = m;
}

// ---------------------------------------------------------------------------
// Host orchestration
// ---------------------------------------------------------------------------
namespace {
struct LayerCfg { int C, Kpad, O; };

inline void run_layer(const _Float16* A, int M, const LayerCfg& L,
                      const float* W, const float* b, const float* g,
                      const float* bt, const float* mu, const float* var,
                      _Float16* Wh, float* scale, float* shift,
                      _Float16* out, hipStream_t stream) {
  const int tot = L.O * L.Kpad;
  prep_layer_kernel<<<(tot + 255) / 256, 256, 0, stream>>>(
      W, b, g, bt, mu, var, L.O, L.C, L.Kpad, Wh, scale, shift);
  if (L.O == 32) {
    dim3 grid(M / 16, L.O / 32);
    const size_t smem = (size_t)2 * 16 * L.Kpad * sizeof(_Float16);
    wmma_gemm_bn_relu<2><<<grid, 32, smem, stream>>>(A, Wh, scale, shift, out,
                                                     M, L.Kpad, L.O);
  } else {
    dim3 grid(M / 16, L.O / 64);
    const size_t smem = (size_t)4 * 16 * L.Kpad * sizeof(_Float16);
    wmma_gemm_bn_relu<4><<<grid, 32, smem, stream>>>(A, Wh, scale, shift, out,
                                                     M, L.Kpad, L.O);
  }
}
} // namespace

extern "C" void kernel_launch(void* const* d_in, const int* in_sizes, int n_in,
                              void* d_out, int out_size, void* d_ws, size_t ws_size,
                              hipStream_t stream) {
  (void)in_sizes; (void)n_in; (void)out_size; (void)ws_size;

  const int B = 8, N1 = 8192, S1 = 512, K1 = 32, S2 = 128, K2 = 16;
  const int rows1 = B * S1 * K1;   // 131072
  const int rows2 = B * S2 * K2;   // 16384
  const int rows3 = B * S2;        // 1024 (group_all)

  const float* xyz = (const float*)d_in[0];
  const float *Wp[9], *bp[9], *gp[9], *btp[9], *mup[9], *varp[9];
  for (int l = 0; l < 9; ++l) {
    Wp[l]   = (const float*)d_in[1 + l * 6 + 0];
    bp[l]   = (const float*)d_in[1 + l * 6 + 1];
    gp[l]   = (const float*)d_in[1 + l * 6 + 2];
    btp[l]  = (const float*)d_in[1 + l * 6 + 3];
    mup[l]  = (const float*)d_in[1 + l * 6 + 4];
    varp[l] = (const float*)d_in[1 + l * 6 + 5];
  }

  const LayerCfg cfg[9] = {
      {3, 32, 32},    {32, 32, 32},   {32, 32, 64},      // sa1
      {67, 96, 64},   {64, 64, 128},  {128, 128, 256},   // sa2
      {259, 288, 256},{256, 256, 512},{512, 512, 1024}}; // sa3

  // bump allocator over workspace
  char* wp = (char*)d_ws;
  auto alloc = [&](size_t bytes) -> void* {
    void* r = (void*)wp;
    wp += (bytes + 255) & ~(size_t)255;
    return r;
  };

  int*      fidx1  = (int*)alloc((size_t)B * S1 * 4);
  float*    nxyz1  = (float*)alloc((size_t)B * S1 * 3 * 4);
  int*      idx1   = (int*)alloc((size_t)B * S1 * K1 * 4);
  int*      fidx2  = (int*)alloc((size_t)B * S2 * 4);
  float*    nxyz2  = (float*)alloc((size_t)B * S2 * 3 * 4);
  int*      idx2   = (int*)alloc((size_t)B * S2 * K2 * 4);
  _Float16* l1_pts = (_Float16*)alloc((size_t)B * S1 * 64 * 2);
  _Float16* l2_pts = (_Float16*)alloc((size_t)B * S2 * 256 * 2);
  _Float16* bufA   = (_Float16*)alloc((size_t)rows1 * 32 * 2);   // 8 MB
  _Float16* bufB   = (_Float16*)alloc((size_t)rows1 * 64 * 2);   // 16 MB
  _Float16* Wh[9]; float* scl[9]; float* shf[9];
  for (int l = 0; l < 9; ++l) {
    Wh[l]  = (_Float16*)alloc((size_t)cfg[l].O * cfg[l].Kpad * 2);
    scl[l] = (float*)alloc((size_t)cfg[l].O * 4);
    shf[l] = (float*)alloc((size_t)cfg[l].O * 4);
  }

  // ---------------- Stage 1: SA(512, r=4, K=32, [32,32,64]) ----------------
  fps_kernel<<<B, 256, 0, stream>>>(xyz, N1, S1, fidx1);
  gather_xyz_kernel<<<(B * S1 + 255) / 256, 256, 0, stream>>>(xyz, fidx1, nxyz1, N1, S1, B);
  ballq_kernel<<<(B * S1 * 32 + 255) / 256, 256, 0, stream>>>(xyz, nxyz1, idx1,
                                                              B, N1, S1, K1, 16.0f);
  {
    const int tot = rows1 * 32;
    group_rel_kernel<<<(tot + 255) / 256, 256, 0, stream>>>(
        xyz, nxyz1, idx1, nullptr, 0, bufA, N1, S1, K1, 32, tot);
  }
  run_layer(bufA, rows1, cfg[0], Wp[0], bp[0], gp[0], btp[0], mup[0], varp[0],
            Wh[0], scl[0], shf[0], bufB, stream);
  run_layer(bufB, rows1, cfg[1], Wp[1], bp[1], gp[1], btp[1], mup[1], varp[1],
            Wh[1], scl[1], shf[1], bufA, stream);
  run_layer(bufA, rows1, cfg[2], Wp[2], bp[2], gp[2], btp[2], mup[2], varp[2],
            Wh[2], scl[2], shf[2], bufB, stream);
  maxpool_kernel<<<(B * S1 * 64 + 255) / 256, 256, 0, stream>>>(
      bufB, K1, 64, B * S1, l1_pts, nullptr);

  // ---------------- Stage 2: SA(128, r=8, K=16, [64,128,256]) --------------
  fps_kernel<<<B, 256, 0, stream>>>(nxyz1, S1, S2, fidx2);
  gather_xyz_kernel<<<(B * S2 + 255) / 256, 256, 0, stream>>>(nxyz1, fidx2, nxyz2, S1, S2, B);
  ballq_kernel<<<(B * S2 * 32 + 255) / 256, 256, 0, stream>>>(nxyz1, nxyz2, idx2,
                                                              B, S1, S2, K2, 64.0f);
  {
    const int tot = rows2 * 96;
    group_rel_kernel<<<(tot + 255) / 256, 256, 0, stream>>>(
        nxyz1, nxyz2, idx2, l1_pts, 64, bufA, S1, S2, K2, 96, tot);
  }
  run_layer(bufA, rows2, cfg[3], Wp[3], bp[3], gp[3], btp[3], mup[3], varp[3],
            Wh[3], scl[3], shf[3], bufB, stream);
  run_layer(bufB, rows2, cfg[4], Wp[4], bp[4], gp[4], btp[4], mup[4], varp[4],
            Wh[4], scl[4], shf[4], bufA, stream);
  run_layer(bufA, rows2, cfg[5], Wp[5], bp[5], gp[5], btp[5], mup[5], varp[5],
            Wh[5], scl[5], shf[5], bufB, stream);
  maxpool_kernel<<<(B * S2 * 256 + 255) / 256, 256, 0, stream>>>(
      bufB, K2, 256, B * S2, l2_pts, nullptr);

  // ---------------- Stage 3: group_all, [256,512,1024] ---------------------
  {
    const int tot = rows3 * 288;
    group_all_kernel<<<(tot + 255) / 256, 256, 0, stream>>>(
        nxyz2, l2_pts, 256, bufA, 288, tot);
  }
  run_layer(bufA, rows3, cfg[6], Wp[6], bp[6], gp[6], btp[6], mup[6], varp[6],
            Wh[6], scl[6], shf[6], bufB, stream);
  run_layer(bufB, rows3, cfg[7], Wp[7], bp[7], gp[7], btp[7], mup[7], varp[7],
            Wh[7], scl[7], shf[7], bufA, stream);
  run_layer(bufA, rows3, cfg[8], Wp[8], bp[8], gp[8], btp[8], mup[8], varp[8],
            Wh[8], scl[8], shf[8], bufB, stream);
  maxpool_kernel<<<(B * 1024 + 255) / 256, 256, 0, stream>>>(
      bufB, S2, 1024, B, nullptr, (float*)d_out);
}
